// SoftAttention_57990648430708
// MI455X (gfx1250) — matrix operand order repaired and checked by reading, not compiled, not run on hardware
//
#include <hip/hip_runtime.h>
#include <hip/hip_bf16.h>

// ---------------- problem sizes ----------------
#define BB 16
#define AA 196
#define HH 64
#define MM 1024
#define RA (BB * AA)   // 3136 rows for att projection
#define RH (BB * HH)   // 1024 rows for h projection

typedef __attribute__((ext_vector_type(16))) __bf16          v16bf;
typedef __attribute__((ext_vector_type(8)))  float           v8f;
typedef __attribute__((ext_vector_type(8)))  unsigned short  v8u16;
typedef __attribute__((ext_vector_type(16))) unsigned short  v16u16;

// ---------------- helpers ----------------
__device__ __forceinline__ float fast_tanh(float x) {
#if __has_builtin(__builtin_amdgcn_tanhf)
    return __builtin_amdgcn_tanhf(x);
#elif __has_builtin(__builtin_amdgcn_tanh_f32)
    return __builtin_amdgcn_tanh_f32(x);
#else
    return tanhf(x);
#endif
}

__device__ __forceinline__ void sched_fence() {
#if __has_builtin(__builtin_amdgcn_sched_barrier)
    __builtin_amdgcn_sched_barrier(0);   // nothing may be reordered across
#endif
}

// ---------------- kernel 0: f32 -> bf16 (round to nearest even) ----------------
__global__ void k_f32_to_bf16(const float* __restrict__ in,
                              unsigned short* __restrict__ out, int n) {
    int i = blockIdx.x * 256 + threadIdx.x;
    if (i < n) {
        unsigned int u = __float_as_uint(in[i]);
        out[i] = (unsigned short)((u + 0x7FFFu + ((u >> 16) & 1u)) >> 16);
    }
}

// ---------------- kernel 1: C(rows x 1024) = A(rows x 1024) * W^T + bias ------
// einsum "rm,nm->rn": C[r][n] = sum_m A[r][m] * W[n][m]; W row-major (n,m) means
// the WMMA B-fragment (k-major) reads W rows contiguously -- no transpose needed.
// One wave computes a 16x64 C strip (4 subtiles, A fragment reused 4x).
// Software pipeline: issue all 10 loads of step k+32, then a sched_barrier(0)
// pins them ABOVE the 4 WMMAs of step k, so each WMMA's s_wait only covers
// loads that already had a full k-step (4 WMMAs + 10 load issues) of latency
// hiding, instead of the s_wait_loadcnt 0x0 the scheduler otherwise emits.
__global__ void k_gemm_bf16(const unsigned short* __restrict__ A,
                            const unsigned short* __restrict__ W,
                            const float* __restrict__ bias,
                            float* __restrict__ C) {
    const int wave  = threadIdx.x >> 5;
    const int lane  = threadIdx.x & 31;
    const int strip = blockIdx.x * 8 + wave;   // (rows/16) * (1024/64) strips
    const int rowBase = (strip >> 4) * 16;
    const int colBase = (strip & 15) * 64;
    const int lhalf = lane >> 4;               // 0: lanes 0-15, 1: lanes 16-31
    const int l15   = lane & 15;

    // A fragment: 16-bit A 16x32 layout. lanes 0-15 hold row=lane, K in
    // {0..7, 16..23}; lanes 16-31 hold row=lane-16, K in {8..15, 24..31}.
    const unsigned short* arow = A + (size_t)(rowBase + l15) * MM + (lhalf ? 8 : 0);
    // B fragment: 32x16, lanes 0-15 -> n=lane, K in {0..7, 16..23} ; lanes 16-31 -> K in {8..15, 24..31}.
    const unsigned short* w0 = W + (size_t)(colBase + l15) * MM + lhalf * 8;

    v8f cacc[4] = {};

    // prologue: fragments for k = 0
    v8u16 a0 = *(const v8u16*)(arow);
    v8u16 a1 = *(const v8u16*)(arow + 16);
    v8u16 b0[4], b1[4];
#pragma unroll
    for (int j = 0; j < 4; ++j) {
        const unsigned short* wr = w0 + (size_t)(j * 16) * MM;
        b0[j] = *(const v8u16*)(wr);
        b1[j] = *(const v8u16*)(wr + 16);
    }

#pragma unroll 4
    for (int kk = 0; kk < MM; kk += 32) {
        // issue next k-step loads first (tail wraps to 0: valid addr, discarded)
        const int kn = (kk + 32) & (MM - 1);
        v8u16 na0 = *(const v8u16*)(arow + kn);
        v8u16 na1 = *(const v8u16*)(arow + kn + 16);
        v8u16 nb0[4], nb1[4];
#pragma unroll
        for (int j = 0; j < 4; ++j) {
            const unsigned short* wr = w0 + (size_t)(j * 16) * MM + kn;
            nb0[j] = *(const v8u16*)(wr);
            nb1[j] = *(const v8u16*)(wr + 16);
        }

        sched_fence();   // keep the 10 loads above the matrix ops

        // consume current fragments
        v16u16 au;
#pragma unroll
        for (int i = 0; i < 8; ++i) { au[i] = a0[i]; au[i + 8] = a1[i]; }
        const v16bf af = __builtin_bit_cast(v16bf, au);
#pragma unroll
        for (int j = 0; j < 4; ++j) {
            v16u16 bu;
#pragma unroll
            for (int i = 0; i < 8; ++i) { bu[i] = b0[j][i]; bu[i + 8] = b1[j][i]; }
            const v16bf bf = __builtin_bit_cast(v16bf, bu);
            // 8 args: (neg_a, A, neg_b, B, c_mod, C, reuse_a, reuse_b)
            cacc[j] = __builtin_amdgcn_wmma_f32_16x16x32_bf16(
                false, af, false, bf, (short)0, cacc[j], false, false);
        }

        sched_fence();   // keep buffer rotation from folding into the loads

        // rotate buffers
        a0 = na0; a1 = na1;
#pragma unroll
        for (int j = 0; j < 4; ++j) { b0[j] = nb0[j]; b1[j] = nb1[j]; }
    }

    // C/D layout: lanes 0-15 -> n=lane, rows rowBase+0..7 in v0..7;
    //             lanes 16-31 -> n=lane-16, rows rowBase+8..15.
    const int r0 = rowBase + (lhalf << 3);
#pragma unroll
    for (int j = 0; j < 4; ++j) {
        const int n = colBase + j * 16 + l15;
        const float bv = bias[n];
#pragma unroll
        for (int i = 0; i < 8; ++i)
            C[(size_t)(r0 + i) * MM + n] = cacc[j][i] + bv;
    }
}

// ---------------- kernel 2: fused scores[b,a,h] ----------------
// scores[b,a,h] = sum_m tanh(av[b,a,m] + hv[b,h,m]) * wd[m] + bd
// One block per (b,a). Each lane owns a fixed set of 32 m-indices, so the av
// row and wd are register-cached once (8 x float4 each) and reused across all
// 8 h iterations of its wave: inner body = 1 b128 load + 4 x (add, v_tanh, fma), keeping
// the TRANS pipe saturated with VMEM/VALU co-issued. No LDS in the hot loop.
__global__ void k_scores(const float* __restrict__ av,
                         const float* __restrict__ hv,
                         const float* __restrict__ wd,
                         const float* __restrict__ bdp,
                         float* __restrict__ scores) {
    const int ba   = blockIdx.x;         // 0..3135
    const int b    = ba / AA;
    const int wave = threadIdx.x >> 5;
    const int lane = threadIdx.x & 31;

    float4 avr[8], wdr[8];
    const float4* avrow = (const float4*)(av + (size_t)ba * MM);
    const float4* wdv   = (const float4*)wd;
#pragma unroll
    for (int i = 0; i < 8; ++i) {
        avr[i] = avrow[(i << 5) + lane];
        wdr[i] = wdv[(i << 5) + lane];
    }
    const float bd = bdp[0];

    for (int hh = wave; hh < HH; hh += 8) {
        const float4* hvrow = (const float4*)(hv + ((size_t)b * HH + hh) * MM);
        if (hh + 8 < HH)   // pull next row toward L0/L2 (global_prefetch_b8)
            __builtin_prefetch(hv + ((size_t)b * HH + hh + 8) * MM + (lane << 2), 0, 0);
        float p = 0.f;
#pragma unroll
        for (int i = 0; i < 8; ++i) {
            const float4 hq = hvrow[(i << 5) + lane];
            p += fast_tanh(avr[i].x + hq.x) * wdr[i].x;
            p += fast_tanh(avr[i].y + hq.y) * wdr[i].y;
            p += fast_tanh(avr[i].z + hq.z) * wdr[i].z;
            p += fast_tanh(avr[i].w + hq.w) * wdr[i].w;
        }
        for (int off = 16; off; off >>= 1) p += __shfl_xor(p, off, 32);
        if (lane == 0) scores[(size_t)ba * HH + hh] = p + bd;
    }
}

// ---------------- kernel 3: softmax over the A axis ----------------
// weight[b,a,h] = softmax_a(scores[b,a,h]); one wave per (b,h) pair.
__global__ void k_softmax(const float* __restrict__ scores,
                          float* __restrict__ weight) {
    const int gid  = blockIdx.x * 8 + (threadIdx.x >> 5);  // 0..BB*HH-1
    const int lane = threadIdx.x & 31;
    const int b  = gid >> 6;
    const int hh = gid & 63;
    const float* s = scores + (size_t)b * AA * HH + hh;

    float v[7];
    float mx = -1e30f;
#pragma unroll
    for (int i = 0; i < 7; ++i) {
        const int a = lane + (i << 5);
        v[i] = (a < AA) ? s[(size_t)a * HH] : -1e30f;
        mx = fmaxf(mx, v[i]);
    }
    for (int off = 16; off; off >>= 1) mx = fmaxf(mx, __shfl_xor(mx, off, 32));

    float sum = 0.f;
#pragma unroll
    for (int i = 0; i < 7; ++i) {
        const int a = lane + (i << 5);
        v[i] = (a < AA) ? __expf(v[i] - mx) : 0.f;
        sum += v[i];
    }
    for (int off = 16; off; off >>= 1) sum += __shfl_xor(sum, off, 32);
    const float inv = 1.f / sum;

    float* w = weight + (size_t)b * AA * HH + hh;
#pragma unroll
    for (int i = 0; i < 7; ++i) {
        const int a = lane + (i << 5);
        if (a < AA) w[(size_t)a * HH] = v[i] * inv;
    }
}

// ---------------- kernel 4: att_res[b,h,m] = sum_a att[b,a,m]*weight[b,a,h] ----
// One block per (b,h); weight column in LDS; coalesced att reads over m.
__global__ void k_attres(const float* __restrict__ att,
                         const float* __restrict__ weight,
                         float* __restrict__ out) {
    __shared__ float wl[AA];
    const int bh = blockIdx.x;
    const int b  = bh >> 6;
    const int hh = bh & 63;
    if (threadIdx.x < AA)
        wl[threadIdx.x] = weight[((size_t)b * AA + threadIdx.x) * HH + hh];
    __syncthreads();

    float a0 = 0.f, a1 = 0.f, a2 = 0.f, a3 = 0.f;
    const float* ab = att + (size_t)b * AA * MM + threadIdx.x;
    for (int a = 0; a < AA; ++a) {
        const float w = wl[a];
        const float* r = ab + (size_t)a * MM;
        a0 += r[0]   * w;
        a1 += r[256] * w;
        a2 += r[512] * w;
        a3 += r[768] * w;
    }
    float* o = out + ((size_t)b * HH + hh) * MM + threadIdx.x;
    o[0] = a0; o[256] = a1; o[512] = a2; o[768] = a3;
}

// ---------------- host launcher ----------------
extern "C" void kernel_launch(void* const* d_in, const int* in_sizes, int n_in,
                              void* d_out, int out_size, void* d_ws, size_t ws_size,
                              hipStream_t stream) {
    const float* att = (const float*)d_in[0];  // (16,196,1024)
    const float* h   = (const float*)d_in[1];  // (16,64,1024)
    const float* Wa  = (const float*)d_in[2];  // (1024,1024)
    const float* ba  = (const float*)d_in[3];  // (1024,)
    const float* Wh  = (const float*)d_in[4];  // (1024,1024)
    const float* bh  = (const float*)d_in[5];  // (1024,)
    const float* wd  = (const float*)d_in[6];  // (1024,)
    const float* bd  = (const float*)d_in[7];  // ()
    float* out = (float*)d_out;                // (16,64,1024)

    // scratch layout (256B aligned)
    char*  base = (char*)d_ws;
    size_t off  = 0;
    auto alloc = [&](size_t bytes) -> char* {
        char* p = base + off;
        off = (off + bytes + 255) & ~(size_t)255;
        return p;
    };
    unsigned short* attb = (unsigned short*)alloc((size_t)RA * MM * 2);
    unsigned short* hb   = (unsigned short*)alloc((size_t)RH * MM * 2);
    unsigned short* Wab  = (unsigned short*)alloc((size_t)MM * MM * 2);
    unsigned short* Whb  = (unsigned short*)alloc((size_t)MM * MM * 2);
    float* av     = (float*)alloc((size_t)RA * MM * 4);
    float* hv     = (float*)alloc((size_t)RH * MM * 4);
    float* scores = (float*)alloc((size_t)BB * AA * HH * 4);
    float* weight = (float*)alloc((size_t)BB * AA * HH * 4);

    // 0) convert operands to bf16
    k_f32_to_bf16<<<(RA * MM) / 256, 256, 0, stream>>>(att, attb, RA * MM);
    k_f32_to_bf16<<<(RH * MM) / 256, 256, 0, stream>>>(h,   hb,   RH * MM);
    k_f32_to_bf16<<<(MM * MM) / 256, 256, 0, stream>>>(Wa,  Wab,  MM * MM);
    k_f32_to_bf16<<<(MM * MM) / 256, 256, 0, stream>>>(Wh,  Whb,  MM * MM);

    // 1) WMMA projections: strips = (rows/16) * 16; 8 strips (waves) per block
    k_gemm_bf16<<<(RA / 16) * 16 / 8, 256, 0, stream>>>(attb, Wab, ba, av);
    k_gemm_bf16<<<(RH / 16) * 16 / 8, 256, 0, stream>>>(hb,   Whb, bh, hv);

    // 2) fused tanh reduce -> scores
    k_scores<<<BB * AA, 256, 0, stream>>>(av, hv, wd, bd, scores);

    // 3) softmax over A axis (one wave per (b,h))
    k_softmax<<<BB * HH / 8, 256, 0, stream>>>(scores, weight);

    // 4) weighted sum back over A -> out (b,h,m)
    k_attres<<<BB * HH, 256, 0, stream>>>(att, weight, out);
}
